// Transformer_P_18485539242600
// MI455X (gfx1250) — compile-verified, matrix operand order checked by reading, not run on hardware
//
#include <hip/hip_runtime.h>
#include <math.h>

#define DIMC 512
#define NHEADS 8
#define DH 64
#define LM 256

typedef __bf16 bf16_t;
typedef __bf16 bf16x8  __attribute__((ext_vector_type(8)));
typedef __bf16 bf16x16 __attribute__((ext_vector_type(16)));
typedef float  f32x8   __attribute__((ext_vector_type(8)));
typedef float  f32x2   __attribute__((ext_vector_type(2)));

union BV16 { bf16x16 v; bf16x8 h[2]; };

static __device__ __forceinline__ bf16x16 mk16(bf16x8 lo, bf16x8 hi) {
  BV16 u; u.h[0] = lo; u.h[1] = hi; return u.v;
}
static __device__ __forceinline__ bf16x16 ld16(const bf16_t* lo, const bf16_t* hi) {
  return mk16(*(const bf16x8*)lo, *(const bf16x8*)hi);
}
static __device__ __forceinline__ float  b2f(bf16_t x) { return (float)x; }
static __device__ __forceinline__ bf16_t f2b(float x)  { return (bf16_t)x; }

static __device__ __forceinline__ f32x8 wmma_bf16(bf16x16 a, bf16x16 b, f32x8 c) {
  return __builtin_amdgcn_wmma_f32_16x16x32_bf16(false, a, false, b, (short)0, c, false, false);
}

// Async copy 16B/lane from global to LDS (CDNA5 GLOBAL_LOAD_ASYNC_TO_LDS_B128, ASYNCcnt)
static __device__ __forceinline__ void async_ld(unsigned ldsoff, const bf16_t* g) {
  asm volatile("global_load_async_to_lds_b128 %0, %1, off"
               :: "v"(ldsoff), "v"((unsigned long long)(uintptr_t)g)
               : "memory");
}

// ---- block reductions (blockDim.x == 256) ----
static __device__ __forceinline__ float block_sum(float* red, int t, float v) {
  red[t] = v; __syncthreads();
  for (int s = 128; s > 0; s >>= 1) { if (t < s) red[t] += red[t + s]; __syncthreads(); }
  float r = red[0]; __syncthreads(); return r;
}
static __device__ __forceinline__ float block_max(float* red, int t, float v) {
  red[t] = v; __syncthreads();
  for (int s = 128; s > 0; s >>= 1) { if (t < s) red[t] = fmaxf(red[t], red[t + s]); __syncthreads(); }
  float r = red[0]; __syncthreads(); return r;
}

// ============================================================
// GEMM: C[M,N] (f32) = alpha * A[M,K](bf16,rowmaj) @ BT[N,K](bf16,rowmaj)^T
// Macro tile 128x64, 8 waves, each wave a 32x32 block (2x2 accumulators).
// K-chunks of 64 double-buffered in LDS via GLOBAL_LOAD_ASYNC_TO_LDS_B128,
// synchronized with s_wait_asynccnt. M mult of 128; N,K mult of 64.
// block=256, grid=(N/64, M/128, batch)
// ============================================================
__global__ void gemm_bt_bf16(float* __restrict__ C, const bf16_t* __restrict__ A,
                             const bf16_t* __restrict__ BT,
                             int N, int K, long sA, long sB, long sC, float alpha)
{
  __shared__ bf16_t lds[2 * (128 + 64) * 72];   // [buf][A 128x72 | B 64x72]
  const int b = blockIdx.z;
  A  += (long)b * sA;  BT += (long)b * sB;  C += (long)b * sC;
  const int n0 = blockIdx.x * 64;
  const int m0 = blockIdx.y * 128;
  const int tid  = threadIdx.x;
  const int lane = tid & 31, wave = tid >> 5;
  const int wm = wave >> 1;        // 0..3  (row block of 32)
  const int wn = wave & 1;         // 0..1  (col block of 32)
  const int l15 = lane & 15;
  const int kboff = (lane >> 4) << 3;   // 0 or 8
  const int rowq = tid >> 3;            // 0..31
  const int colq = (tid & 7) * 8;       // 0..56 (halfs)

  const unsigned ldsbase = (unsigned)(uintptr_t)(&lds[0]);
  const unsigned BUFH = (128 + 64) * 72;      // halfs per buffer

  long aoff[4], boff[2];
  unsigned aldso[4], bldso[2];
#pragma unroll
  for (int q = 0; q < 4; ++q) {
    aoff[q]  = (long)(m0 + q * 32 + rowq) * K + colq;
    aldso[q] = (unsigned)(((q * 32 + rowq) * 72 + colq) * 2);
  }
#pragma unroll
  for (int q = 0; q < 2; ++q) {
    boff[q]  = (long)(n0 + q * 32 + rowq) * K + colq;
    bldso[q] = (unsigned)(((128 + q * 32 + rowq) * 72 + colq) * 2);
  }

  f32x8 acc[2][2] = {};
  const int nch = K >> 6;

  // prologue: chunk 0 -> buffer 0
#pragma unroll
  for (int q = 0; q < 4; ++q) async_ld(ldsbase + aldso[q], A + aoff[q]);
#pragma unroll
  for (int q = 0; q < 2; ++q) async_ld(ldsbase + bldso[q], BT + boff[q]);

  for (int i = 0; i < nch; ++i) {
    const int cur = i & 1;
    if (i + 1 < nch) {
      const unsigned nb = ldsbase + (unsigned)(cur ^ 1) * BUFH * 2;
      const long kk = (long)(i + 1) << 6;
#pragma unroll
      for (int q = 0; q < 4; ++q) async_ld(nb + aldso[q], A + aoff[q] + kk);
#pragma unroll
      for (int q = 0; q < 2; ++q) async_ld(nb + bldso[q], BT + boff[q] + kk);
      asm volatile("s_wait_asynccnt 0x6" ::: "memory");  // chunk i landed
    } else {
      asm volatile("s_wait_asynccnt 0x0" ::: "memory");
    }
    __syncthreads();
    const bf16_t* sA = lds + (size_t)cur * BUFH;
    const bf16_t* sB = sA + 128 * 72;
#pragma unroll
    for (int k2 = 0; k2 < 64; k2 += 32) {
      const int kb = k2 + kboff;
      const bf16_t* ar0 = sA + (wm * 32 + l15) * 72 + kb;
      const bf16_t* ar1 = sA + (wm * 32 + 16 + l15) * 72 + kb;
      const bf16_t* br0 = sB + (wn * 32 + l15) * 72 + kb;
      const bf16_t* br1 = sB + (wn * 32 + 16 + l15) * 72 + kb;
      bf16x16 a0v = ld16(ar0, ar0 + 16);
      bf16x16 a1v = ld16(ar1, ar1 + 16);
      bf16x16 b0v = ld16(br0, br0 + 16);
      bf16x16 b1v = ld16(br1, br1 + 16);
      acc[0][0] = wmma_bf16(a0v, b0v, acc[0][0]);
      acc[0][1] = wmma_bf16(a0v, b1v, acc[0][1]);
      acc[1][0] = wmma_bf16(a1v, b0v, acc[1][0]);
      acc[1][1] = wmma_bf16(a1v, b1v, acc[1][1]);
    }
    __syncthreads();
  }

  const int rhi = (lane >> 4) * 8;
#pragma unroll
  for (int si = 0; si < 2; ++si)
#pragma unroll
    for (int sj = 0; sj < 2; ++sj)
#pragma unroll
      for (int r = 0; r < 8; ++r)
        C[(long)(m0 + wm * 32 + si * 16 + r + rhi) * N + n0 + wn * 32 + sj * 16 + l15]
            = alpha * acc[si][sj][r];
}

// ============================================================
// f32 WMMA GEMM: C[M,N] = alpha * A[M,K] @ B[K,N]   (all f32)
// v_wmma_f32_16x16x4_f32, wave per 16x16 tile. block=128, grid=(tiles/4,1,batch)
// ============================================================
__global__ void gemm_nn_f32(float* __restrict__ C, const float* __restrict__ A,
                            const float* __restrict__ B,
                            int M, int N, int K, long sA, long sB, long sC, float alpha)
{
  const int b = blockIdx.z;
  A += (long)b * sA; B += (long)b * sB; C += (long)b * sC;
  const int lane = threadIdx.x & 31, wave = threadIdx.x >> 5;
  const int ntiles = N >> 4;
  const int tile = blockIdx.x * (blockDim.x >> 5) + wave;
  if (tile >= (M >> 4) * ntiles) return; // wave-uniform
  const int m0 = (tile / ntiles) * 16;
  const int n0 = (tile % ntiles) * 16;
  const int l15 = lane & 15;
  const int kb0 = (lane >> 4) * 2;
  f32x8 c = {};
  for (int k = 0; k < K; k += 4) {
    f32x2 a = *(const f32x2*)(A + (long)(m0 + l15) * K + k + kb0);
    f32x2 bb;
    bb[0] = B[(long)(k + kb0) * N + n0 + l15];
    bb[1] = B[(long)(k + kb0 + 1) * N + n0 + l15];
    c = __builtin_amdgcn_wmma_f32_16x16x4_f32(false, a, false, bb, (short)0, c, false, false);
  }
  const int rhi = (lane >> 4) * 8;
#pragma unroll
  for (int r = 0; r < 8; ++r)
    C[(long)(m0 + r + rhi) * N + n0 + l15] = alpha * c[r];
}

// ============================================================
// Fused sim1 path: out[h,i,:] = softmax(q[h,i,:] @ k_land[h]^T) @ Wh[h]
// WhT is (8, 64, 256) bf16. block=128 (4 waves), grid=(P/16, 8)
// ============================================================
__global__ void attn_rows(float* __restrict__ attnO, const bf16_t* __restrict__ qh,
                          const bf16_t* __restrict__ kland, const bf16_t* __restrict__ WhT,
                          int P)
{
  __shared__ float  logit[16][256];
  __shared__ bf16_t probs[16][256];
  const int h  = blockIdx.y;
  const int i0 = blockIdx.x * 16;
  const int lane = threadIdx.x & 31, wave = threadIdx.x >> 5;
  const int l15 = lane & 15;
  const int kboff = (lane >> 4) * 8;

  // phase 1: logits = q(16x64) @ k_land^T (256x64)^T
  const bf16_t* qrow = qh + ((long)h * P + i0 + l15) * DH;
  bf16x16 a0 = ld16(qrow + kboff,      qrow + kboff + 16);
  bf16x16 a1 = ld16(qrow + 32 + kboff, qrow + 48 + kboff);
#pragma unroll
  for (int t = 0; t < 4; ++t) {
    const int n0 = (wave * 4 + t) * 16;
    const bf16_t* krow = kland + ((long)h * LM + n0 + l15) * DH;
    bf16x16 b0 = ld16(krow + kboff,      krow + kboff + 16);
    bf16x16 b1 = ld16(krow + 32 + kboff, krow + 48 + kboff);
    f32x8 c = {};
    c = wmma_bf16(a0, b0, c);
    c = wmma_bf16(a1, b1, c);
#pragma unroll
    for (int r = 0; r < 8; ++r)
      logit[r + ((lane >> 4) * 8)][n0 + l15] = c[r];
  }
  __syncthreads();

  // phase 2: softmax per row (threads 0..15)
  if (threadIdx.x < 16) {
    const int r = threadIdx.x;
    float mx = -3.4e38f;
    for (int j = 0; j < 256; ++j) mx = fmaxf(mx, logit[r][j]);
    float s = 0.f;
    for (int j = 0; j < 256; ++j) s += __expf(logit[r][j] - mx);
    const float inv = 1.f / s;
    for (int j = 0; j < 256; ++j) probs[r][j] = f2b(__expf(logit[r][j] - mx) * inv);
  }
  __syncthreads();

  // phase 3: out(16x64) = probs(16x256) @ WhT(64x256)^T
  const int n0 = wave * 16;
  f32x8 c = {};
  for (int kk = 0; kk < 256; kk += 32) {
    bf16x16 a = ld16(&probs[l15][kk + kboff], &probs[l15][kk + kboff + 16]);
    const bf16_t* wrow = WhT + ((long)h * DH + n0 + l15) * 256 + kk;
    bf16x16 b = ld16(wrow + kboff, wrow + kboff + 16);
    c = wmma_bf16(a, b, c);
  }
#pragma unroll
  for (int r = 0; r < 8; ++r)
    attnO[((long)h * P + i0 + r + ((lane >> 4) * 8)) * DH + n0 + l15] = c[r];
}

// ============================================================ elementwise kernels
__global__ void build_h(float* h, const float* feats, const float* cls, long N, long S)
{
  long idx = (long)blockIdx.x * 256 + threadIdx.x;
  if (idx >= S * DIMC) return;
  long r = idx >> 9; int c = idx & 511;
  float v;
  if (r == 0) v = cls[c];
  else { long f = r - 1; if (f >= N) f -= N; v = feats[f * DIMC + c]; }
  h[idx] = v;
}

__global__ void ln_pad(const float* __restrict__ h, const float* __restrict__ g,
                       const float* __restrict__ bb, bf16_t* __restrict__ Xb, int S, int pad)
{
  __shared__ float red[256];
  const int r = blockIdx.x, t = threadIdx.x;
  if (r < pad) { Xb[(long)r * DIMC + t] = f2b(0.f); Xb[(long)r * DIMC + t + 256] = f2b(0.f); return; }
  const float* x = h + (long)(r - pad) * DIMC;
  float v0 = x[t], v1 = x[t + 256];
  float mu = block_sum(red, t, v0 + v1) * (1.f / DIMC);
  float d0 = v0 - mu, d1 = v1 - mu;
  float var = block_sum(red, t, d0 * d0 + d1 * d1) * (1.f / DIMC);
  float rs = rsqrtf(var + 1e-5f);
  Xb[(long)r * DIMC + t]       = f2b(d0 * rs * g[t] + bb[t]);
  Xb[(long)r * DIMC + t + 256] = f2b(d1 * rs * g[t + 256] + bb[t + 256]);
}

__global__ void final_ln(float* __restrict__ out, const float* __restrict__ h,
                         const float* __restrict__ g, const float* __restrict__ bb)
{
  __shared__ float red[256];
  const int r = blockIdx.x, t = threadIdx.x;
  const float* x = h + (long)r * DIMC;
  float v0 = x[t], v1 = x[t + 256];
  float mu = block_sum(red, t, v0 + v1) * (1.f / DIMC);
  float d0 = v0 - mu, d1 = v1 - mu;
  float var = block_sum(red, t, d0 * d0 + d1 * d1) * (1.f / DIMC);
  float rs = rsqrtf(var + 1e-5f);
  out[(long)r * DIMC + t]       = d0 * rs * g[t] + bb[t];
  out[(long)r * DIMC + t + 256] = d1 * rs * g[t + 256] + bb[t + 256];
}

__global__ void transpose_f32_to_bf16(bf16_t* dst, const float* src, int K, int N)
{ // src: K x N -> dst: N x K
  long idx = (long)blockIdx.x * 256 + threadIdx.x;
  if (idx >= (long)K * N) return;
  int n = (int)(idx % N); long k = idx / N;
  dst[(long)n * K + k] = f2b(src[idx]);
}

__global__ void split_qkv(const float* __restrict__ QKV, bf16_t* qh, bf16_t* kh, bf16_t* vT, int P)
{
  long idx = (long)blockIdx.x * 256 + threadIdx.x; // P*512
  if (idx >= (long)P * DIMC) return;
  int d = idx & 63; int h = (int)((idx >> 6) & 7); long i = idx >> 9;
  const float* row = QKV + i * (3 * DIMC);
  qh[((long)h * P + i) * DH + d] = f2b(row[h * DH + d] * 0.125f);  // 64^-0.5
  kh[((long)h * P + i) * DH + d] = f2b(row[DIMC + h * DH + d]);
  vT[((long)(h * DH + d)) * P + i] = f2b(row[2 * DIMC + h * DH + d]);
}

__global__ void landmarks(const bf16_t* __restrict__ src, bf16_t* __restrict__ dst, int P, int L)
{
  long idx = (long)blockIdx.x * 256 + threadIdx.x; // 8*256*64
  if (idx >= (long)NHEADS * LM * DH) return;
  int d = idx & 63; int j = (int)((idx >> 6) & 255); int h = (int)(idx >> 14);
  const bf16_t* base = src + ((long)h * P + (long)j * L) * DH + d;
  float s = 0.f;
  for (int t = 0; t < L; ++t) s += b2f(base[(long)t * DH]);
  dst[idx] = f2b(s / (float)L);
}

__global__ void softmax_rows(float* __restrict__ X, bf16_t* __restrict__ Xb, int cols)
{
  __shared__ float red[256];
  const long row = blockIdx.x; const int t = threadIdx.x;
  float* x = X + row * (long)cols;
  float mx = -3.4e38f;
  for (int j = t; j < cols; j += 256) mx = fmaxf(mx, x[j]);
  mx = block_max(red, t, mx);
  float s = 0.f;
  for (int j = t; j < cols; j += 256) s += __expf(x[j] - mx);
  s = block_sum(red, t, s);
  const float inv = 1.f / s;
  for (int j = t; j < cols; j += 256) {
    float e = __expf(x[j] - mx) * inv;
    x[j] = e;
    if (Xb) Xb[row * (long)cols + j] = f2b(e);
  }
}

__global__ void abs_rowsum(const float* __restrict__ a2, float* __restrict__ out)
{ // grid 2048 rows
  __shared__ float red[256];
  const long row = blockIdx.x; const int t = threadIdx.x;
  float s = block_sum(red, t, fabsf(a2[row * LM + t]));
  if (t == 0) out[row] = s;
}
__global__ void abs_colsum(const float* __restrict__ a2, float* __restrict__ out)
{ // grid 2048: h=blk>>8, j=blk&255
  __shared__ float red[256];
  const int h = blockIdx.x >> 8, j = blockIdx.x & 255; const int t = threadIdx.x;
  float s = block_sum(red, t, fabsf(a2[((long)h * LM + t) * LM + j]));
  if (t == 0) out[blockIdx.x] = s;
}
__global__ void pinv_scale(const float* rs, const float* cs, float* scal)
{
  __shared__ float red[256];
  const int t = threadIdx.x;
  float mr = 0.f, mc = 0.f;
  for (int j = t; j < NHEADS * LM; j += 256) { mr = fmaxf(mr, rs[j]); mc = fmaxf(mc, cs[j]); }
  mr = block_max(red, t, mr);
  mc = block_max(red, t, mc);
  if (t == 0) scal[0] = 1.f / (mr * mc);
}
__global__ void zinit(float* __restrict__ z, const float* __restrict__ a2, const float* scal)
{
  long idx = (long)blockIdx.x * 256 + threadIdx.x; // 8*256*256
  long h = idx >> 16; int i = (int)((idx >> 8) & 255); int j = (int)(idx & 255);
  z[idx] = a2[(h << 16) + ((long)j << 8) + i] * scal[0];
}
__global__ void ci_minus(float* __restrict__ dst, const float* __restrict__ src, float cval)
{
  long idx = (long)blockIdx.x * 256 + threadIdx.x; // 8*256*256
  int i = (int)((idx >> 8) & 255); int j = (int)(idx & 255);
  dst[idx] = ((i == j) ? cval : 0.f) - src[idx];
}
__global__ void transpose_wh(bf16_t* __restrict__ WhT, const float* __restrict__ Wh)
{
  long idx = (long)blockIdx.x * 256 + threadIdx.x; // 8*256*64
  if (idx >= (long)NHEADS * LM * DH) return;
  int h = (int)(idx >> 14); int j = (int)((idx >> 6) & 255); int d = (int)(idx & 63);
  WhT[((long)(h * DH + d)) * LM + j] = f2b(Wh[idx]);
}
__global__ void conv_res(float* __restrict__ attnO, const bf16_t* __restrict__ vT,
                         const float* __restrict__ rw, int P)
{
  long idx = (long)blockIdx.x * 256 + threadIdx.x; // 8*P*64 = ((h*P+i)*64+d)
  if (idx >= (long)NHEADS * P * DH) return;
  int d = (int)(idx & 63); long hp = idx >> 6; long i = hp % P; int h = (int)(hp / P);
  const bf16_t* vrow = vT + ((long)(h * DH + d)) * P;
  const float* w = rw + h * 33;
  float s = 0.f;
#pragma unroll
  for (int t = 0; t < 33; ++t) {
    long p = i + t - 16;
    if ((unsigned long)p < (unsigned long)P) s += w[t] * b2f(vrow[p]);
  }
  attnO[idx] += s;
}
__global__ void merge_heads(bf16_t* __restrict__ Ob, const float* __restrict__ attnO, int P)
{
  long idx = (long)blockIdx.x * 256 + threadIdx.x; // P*512
  if (idx >= (long)P * DIMC) return;
  int c = (int)(idx & 511); long i = idx >> 9; int h = c >> 6; int d = c & 63;
  Ob[idx] = f2b(attnO[((long)h * P + i) * DH + d]);
}
__global__ void residual(float* __restrict__ hdst, const float* __restrict__ hsrc,
                         const float* __restrict__ Y, const float* __restrict__ ob,
                         long S, int pad)
{
  long idx = (long)blockIdx.x * 256 + threadIdx.x; // S*512
  if (idx >= S * DIMC) return;
  int c = (int)(idx & 511); long s = idx >> 9;
  hdst[idx] = hsrc[idx] + Y[(long)(pad + s) * DIMC + c] + ob[c];
}
__global__ void copy_row512(float* dst, const float* src)
{
  int c = blockIdx.x * 256 + threadIdx.x;
  if (c < DIMC) dst[c] = src[c];
}
__global__ void ppeg_kernel(float* __restrict__ dst, const float* __restrict__ src,
                            const float* __restrict__ w7, const float* __restrict__ b7,
                            const float* __restrict__ w5, const float* __restrict__ b5,
                            const float* __restrict__ w3, const float* __restrict__ b3, int Hd)
{
  long total = (long)Hd * Hd * DIMC;
  long idx = (long)blockIdx.x * 256 + threadIdx.x;
  if (idx >= total) return;
  int c = (int)(idx & 511); long p = idx >> 9;
  int x = (int)(p % Hd), y = (int)(p / Hd);
  const float* W7 = w7 + (long)c * 49;
  const float* W5 = w5 + (long)c * 25;
  const float* W3 = w3 + (long)c * 9;
  float acc = src[(1 + p) * DIMC + c] + b7[c] + b5[c] + b3[c];
  for (int dy = 0; dy < 7; ++dy) { int yy = y + dy - 3; if ((unsigned)yy >= (unsigned)Hd) continue;
    for (int dx = 0; dx < 7; ++dx) { int xx = x + dx - 3; if ((unsigned)xx >= (unsigned)Hd) continue;
      acc += W7[dy * 7 + dx] * src[(1 + (long)yy * Hd + xx) * DIMC + c]; } }
  for (int dy = 0; dy < 5; ++dy) { int yy = y + dy - 2; if ((unsigned)yy >= (unsigned)Hd) continue;
    for (int dx = 0; dx < 5; ++dx) { int xx = x + dx - 2; if ((unsigned)xx >= (unsigned)Hd) continue;
      acc += W5[dy * 5 + dx] * src[(1 + (long)yy * Hd + xx) * DIMC + c]; } }
  for (int dy = 0; dy < 3; ++dy) { int yy = y + dy - 1; if ((unsigned)yy >= (unsigned)Hd) continue;
    for (int dx = 0; dx < 3; ++dx) { int xx = x + dx - 1; if ((unsigned)xx >= (unsigned)Hd) continue;
      acc += W3[dy * 3 + dx] * src[(1 + (long)yy * Hd + xx) * DIMC + c]; } }
  dst[(1 + p) * DIMC + c] = acc;
}

// ============================================================ host
extern "C" void kernel_launch(void* const* d_in, const int* in_sizes, int n_in,
                              void* d_out, int out_size, void* d_ws, size_t ws_size,
                              hipStream_t stream)
{
  const float* feats = (const float*)d_in[0];
  const float* cls   = (const float*)d_in[1];
  const long N   = in_sizes[0] / DIMC;
  int Hd = (int)ceil(sqrt((double)N));
  const long S   = 1 + (long)Hd * Hd;
  const int pad  = (int)((LM - (S % LM)) % LM);
  const long P   = S + pad;          // multiple of 256
  const int  L   = (int)(P / LM);

  size_t off = 0;
  auto alloc = [&](size_t bytes) -> void* {
    off = (off + 255) & ~(size_t)255;
    void* p = (char*)d_ws + off; off += bytes; return p;
  };
  float*  hA    = (float*)alloc((size_t)S * DIMC * 4);
  float*  hB    = (float*)alloc((size_t)S * DIMC * 4);
  bf16_t* Xb    = (bf16_t*)alloc((size_t)P * DIMC * 2);
  bf16_t* WqkvT = (bf16_t*)alloc((size_t)3 * DIMC * DIMC * 2);
  bf16_t* WoutT = (bf16_t*)alloc((size_t)DIMC * DIMC * 2);
  float*  QKV   = (float*)alloc((size_t)P * 3 * DIMC * 4);
  // aliases inside the (dead-after-split) QKV region:
  float*  attnO = QKV;                                           // 8*P*64 f32
  float*  Ybuf  = (float*)((char*)QKV + (size_t)P * DIMC * 4);   // P*512 f32
  bf16_t* Ob    = (bf16_t*)((char*)QKV + (size_t)P * DIMC * 8);  // P*512 bf16
  bf16_t* qh    = (bf16_t*)alloc((size_t)NHEADS * P * DH * 2);
  bf16_t* kh    = (bf16_t*)alloc((size_t)NHEADS * P * DH * 2);
  bf16_t* vT    = (bf16_t*)alloc((size_t)NHEADS * DH * P * 2);
  bf16_t* qland = (bf16_t*)alloc((size_t)NHEADS * LM * DH * 2);
  bf16_t* kland = (bf16_t*)alloc((size_t)NHEADS * LM * DH * 2);
  float*  a2    = (float*)alloc((size_t)NHEADS * LM * LM * 4);
  float*  zA    = (float*)alloc((size_t)NHEADS * LM * LM * 4);
  float*  zB    = (float*)alloc((size_t)NHEADS * LM * LM * 4);
  float*  xz    = (float*)alloc((size_t)NHEADS * LM * LM * 4);
  float*  wA    = (float*)alloc((size_t)NHEADS * LM * LM * 4);
  float*  wB    = (float*)alloc((size_t)NHEADS * LM * LM * 4);
  float*  rsum  = (float*)alloc(NHEADS * LM * 4);
  float*  csum  = (float*)alloc(NHEADS * LM * 4);
  float*  scal  = (float*)alloc(256 * 4);
  float*  S3    = (float*)alloc((size_t)LM * P * 4);
  bf16_t* S3b   = (bf16_t*)alloc((size_t)LM * P * 2);
  float*  a3v   = (float*)alloc((size_t)NHEADS * LM * DH * 4);
  float*  Whb   = (float*)alloc((size_t)NHEADS * LM * DH * 4);
  bf16_t* WhT   = (bf16_t*)alloc((size_t)NHEADS * DH * LM * 2);

  const long eHW   = (long)NHEADS * LM * DH;        // 131072
  const long eHMM  = (long)NHEADS * LM * LM;        // 524288

  auto layer = [&](const float* lng, const float* lnb, const float* qkvw,
                   const float* outw, const float* outb, const float* resw,
                   const float* hsrc, float* hdst)
  {
    transpose_f32_to_bf16<<<(int)(((long)DIMC * 3 * DIMC + 255) / 256), 256, 0, stream>>>(WqkvT, qkvw, DIMC, 3 * DIMC);
    transpose_f32_to_bf16<<<(int)(((long)DIMC * DIMC + 255) / 256), 256, 0, stream>>>(WoutT, outw, DIMC, DIMC);
    ln_pad<<<(int)P, 256, 0, stream>>>(hsrc, lng, lnb, Xb, (int)S, pad);
    gemm_bt_bf16<<<dim3(3 * DIMC / 64, (int)(P / 128), 1), 256, 0, stream>>>(QKV, Xb, WqkvT, 3 * DIMC, DIMC, 0, 0, 0, 1.f);
    split_qkv<<<(int)((P * DIMC + 255) / 256), 256, 0, stream>>>(QKV, qh, kh, vT, (int)P);
    landmarks<<<(int)((eHW + 255) / 256), 256, 0, stream>>>(qh, qland, (int)P, L);
    landmarks<<<(int)((eHW + 255) / 256), 256, 0, stream>>>(kh, kland, (int)P, L);
    // sim2 = q_land @ k_land^T (batched over heads)
    gemm_bt_bf16<<<dim3(LM / 64, LM / 128, NHEADS), 256, 0, stream>>>(a2, qland, kland, LM, DH, LM * DH, LM * DH, LM * LM, 1.f);
    softmax_rows<<<NHEADS * LM, 256, 0, stream>>>(a2, (bf16_t*)nullptr, LM);
    // pinv init
    abs_rowsum<<<NHEADS * LM, 256, 0, stream>>>(a2, rsum);
    abs_colsum<<<NHEADS * LM, 256, 0, stream>>>(a2, csum);
    pinv_scale<<<1, 256, 0, stream>>>(rsum, csum, scal);
    zinit<<<(int)(eHMM / 256), 256, 0, stream>>>(zA, a2, scal);
    float *za = zA, *zb = zB;
    for (int it = 0; it < 6; ++it) {
      gemm_nn_f32<<<dim3(64, 1, NHEADS), 128, 0, stream>>>(xz, a2, za, LM, LM, LM, LM * LM, LM * LM, LM * LM, 1.f);
      ci_minus<<<(int)(eHMM / 256), 256, 0, stream>>>(wA, xz, 7.f);
      gemm_nn_f32<<<dim3(64, 1, NHEADS), 128, 0, stream>>>(wB, xz, wA, LM, LM, LM, LM * LM, LM * LM, LM * LM, 1.f);
      ci_minus<<<(int)(eHMM / 256), 256, 0, stream>>>(wB, wB, 15.f);
      gemm_nn_f32<<<dim3(64, 1, NHEADS), 128, 0, stream>>>(wA, xz, wB, LM, LM, LM, LM * LM, LM * LM, LM * LM, 1.f);
      ci_minus<<<(int)(eHMM / 256), 256, 0, stream>>>(wA, wA, 13.f);
      gemm_nn_f32<<<dim3(64, 1, NHEADS), 128, 0, stream>>>(zb, za, wA, LM, LM, LM, LM * LM, LM * LM, LM * LM, 0.25f);
      float* t = za; za = zb; zb = t;
    }
    // sim3 path per head: S3 = q_land @ k^T ; softmax ; a3v = S3 @ v
    for (int h = 0; h < NHEADS; ++h) {
      gemm_bt_bf16<<<dim3((int)(P / 64), LM / 128, 1), 256, 0, stream>>>(S3, qland + (long)h * LM * DH, kh + (long)h * P * DH, (int)P, DH, 0, 0, 0, 1.f);
      softmax_rows<<<LM, 256, 0, stream>>>(S3, S3b, (int)P);
      gemm_bt_bf16<<<dim3(1, LM / 128, 1), 256, 0, stream>>>(a3v + (long)h * LM * DH, S3b, vT + (long)h * DH * P, DH, (int)P, 0, 0, 0, 1.f);
    }
    // Wh = a2inv @ a3v ; transpose to bf16
    gemm_nn_f32<<<dim3(16, 1, NHEADS), 128, 0, stream>>>(Whb, za, a3v, LM, DH, LM, LM * LM, LM * DH, LM * DH, 1.f);
    transpose_wh<<<(int)((eHW + 255) / 256), 256, 0, stream>>>(WhT, Whb);
    // fused sim1 + combine
    attn_rows<<<dim3((int)(P / 16), NHEADS), 128, 0, stream>>>(attnO, qh, kland, WhT, (int)P);
    conv_res<<<(int)((P * DIMC + 255) / 256), 256, 0, stream>>>(attnO, vT, resw, (int)P);
    merge_heads<<<(int)((P * DIMC + 255) / 256), 256, 0, stream>>>(Ob, attnO, (int)P);
    gemm_bt_bf16<<<dim3(DIMC / 64, (int)(P / 128), 1), 256, 0, stream>>>(Ybuf, Ob, WoutT, DIMC, DIMC, 0, 0, 0, 1.f);
    residual<<<(int)((S * DIMC + 255) / 256), 256, 0, stream>>>(hdst, hsrc, Ybuf, outb, S, pad);
  };

  build_h<<<(int)((S * DIMC + 255) / 256), 256, 0, stream>>>(hA, feats, cls, N, S);

  layer((const float*)d_in[2], (const float*)d_in[3], (const float*)d_in[4],
        (const float*)d_in[5], (const float*)d_in[6], (const float*)d_in[7], hA, hB);

  ppeg_kernel<<<(int)(((long)Hd * Hd * DIMC + 255) / 256), 256, 0, stream>>>(
      hA, hB, (const float*)d_in[8], (const float*)d_in[9], (const float*)d_in[10],
      (const float*)d_in[11], (const float*)d_in[12], (const float*)d_in[13], Hd);
  copy_row512<<<2, 256, 0, stream>>>(hA, hB);

  layer((const float*)d_in[14], (const float*)d_in[15], (const float*)d_in[16],
        (const float*)d_in[17], (const float*)d_in[18], (const float*)d_in[19], hA, hB);

  final_ln<<<(int)S, 256, 0, stream>>>((float*)d_out, hB, (const float*)d_in[20], (const float*)d_in[21]);
}
